// DecoderAttentionLayer_64037962384133
// MI455X (gfx1250) — compile-verified, matrix operand order checked by reading, not run on hardware
//
#include <hip/hip_runtime.h>

// Problem constants (N=2, T=4096, D=1024, HD=64, NH=16, W=128)
#define NB     2
#define TT     4096
#define DDIM   1024
#define HDIM   64
#define NHEAD  16
#define WW     128
#define CCHUNK (TT / WW)
#define MROWS  (NB * TT)
#define EPSF   1.1920929e-07f

typedef __bf16 bf16;
typedef __attribute__((ext_vector_type(16))) __bf16 v16bf;
typedef __attribute__((ext_vector_type(8)))  float  v8f;

union FragB { v16bf v; uint4 q[2]; };

__device__ __forceinline__ bf16 to_bf16(float f) {
    unsigned u = __float_as_uint(f);
    u += 0x7FFFu + ((u >> 16) & 1u);           // round-to-nearest-even
    unsigned short h = (unsigned short)(u >> 16);
    return __builtin_bit_cast(__bf16, h);
}

// ---------------------------------------------------------------------------
// Weight conversion: qkv_weight -> bf16 ; (o_weight * o_scale[row]) -> bf16
// ---------------------------------------------------------------------------
__global__ __launch_bounds__(256) void convw_kernel(
    const float* __restrict__ qw, const float* __restrict__ ow,
    const float* __restrict__ osc, bf16* __restrict__ Wq, bf16* __restrict__ W2)
{
    const int NQ = 3 * DDIM * DDIM;
    int i = blockIdx.x * 256 + threadIdx.x;
    if (i < NQ) {
        Wq[i] = to_bf16(qw[i]);
    } else {
        int j = i - NQ;
        if (j < DDIM * DDIM) W2[j] = to_bf16(ow[j] * osc[j / DDIM]);
    }
}

// ---------------------------------------------------------------------------
// Input RMSNorm -> bf16 activations, one block per row of 1024
// ---------------------------------------------------------------------------
__global__ __launch_bounds__(256) void rmsnorm_kernel(
    const float* __restrict__ x, bf16* __restrict__ xn)
{
    __shared__ float buf[256];
    const int row = blockIdx.x;
    const float* xr = x + (size_t)row * DDIM;
    float ss = 0.f;
    for (int i = threadIdx.x; i < DDIM; i += 256) { float v = xr[i]; ss += v * v; }
    buf[threadIdx.x] = ss;
    __syncthreads();
    for (int s = 128; s > 0; s >>= 1) {
        if (threadIdx.x < s) buf[threadIdx.x] += buf[threadIdx.x + s];
        __syncthreads();
    }
    float scale = rsqrtf(buf[0] / (float)DDIM + EPSF);
    for (int i = threadIdx.x; i < DDIM; i += 256)
        xn[(size_t)row * DDIM + i] = to_bf16(xr[i] * scale);
}

// ---------------------------------------------------------------------------
// LDS-staged bf16 GEMM: C[M,Nout] = A[M,K] * B[Nout,K]^T (+ resid), f32 out.
// Block = 8 waves; block tile 128(M) x 256(N); K staged in chunks of 64
// through LDS (A 16KB + B 32KB, rows padded to 144B to spread banks).
// Each wave computes a 64x64 macro-tile = 4x4 WMMA tiles: per 32-K step
// 8 fragment reads (16x ds_load_b128) feed 16 WMMAs.
// ---------------------------------------------------------------------------
#define KC    64
#define LPAD  8
__global__ __launch_bounds__(256) void gemm_bf16_kernel(
    const bf16* __restrict__ A, const bf16* __restrict__ B,
    float* __restrict__ C, const float* __restrict__ resid,
    int M, int Nout, int K)
{
    __shared__ bf16 sA[128][KC + LPAD];
    __shared__ bf16 sB[256][KC + LPAD];

    const int tid   = threadIdx.x;
    const int lane  = tid & 31;
    const int wave  = tid >> 5;              // 0..7
    const int half  = lane >> 4;
    const int lx    = lane & 15;
    const int waveM = wave >> 2;             // 0..1
    const int waveN = wave & 3;              // 0..3
    const int blockM = blockIdx.y * 128;
    const int blockN = blockIdx.x * 256;

    v8f acc[4][4] = {};

    for (int k0 = 0; k0 < K; k0 += KC) {
        __syncthreads();
        // ---- stage A panel: 128 rows x 64 k (2 threads/row, 64B each) ----
        {
            const int row  = tid >> 1;
            const int part = tid & 1;
            const uint4* src = (const uint4*)(A + (size_t)(blockM + row) * K + k0 + part * 32);
            uint4* dst = (uint4*)(&sA[row][part * 32]);
#pragma unroll
            for (int i = 0; i < 4; ++i) dst[i] = src[i];
        }
        // ---- stage B panel: 256 rows x 64 k (1 thread/row, 128B each) ----
        {
            const uint4* src = (const uint4*)(B + (size_t)(blockN + tid) * K + k0);
            uint4* dst = (uint4*)(&sB[tid][0]);
#pragma unroll
            for (int i = 0; i < 8; ++i) dst[i] = src[i];
        }
        __syncthreads();

#pragma unroll
        for (int ks = 0; ks < KC / 32; ++ks) {
            const int kb = ks * 32 + half * 8;
            FragB a[4], b[4];
#pragma unroll
            for (int i = 0; i < 4; ++i) {
                a[i].q[0] = *(const uint4*)(&sA[waveM * 64 + i * 16 + lx][kb]);
                a[i].q[1] = *(const uint4*)(&sA[waveM * 64 + i * 16 + lx][kb + 16]);
                b[i].q[0] = *(const uint4*)(&sB[waveN * 64 + i * 16 + lx][kb]);
                b[i].q[1] = *(const uint4*)(&sB[waveN * 64 + i * 16 + lx][kb + 16]);
            }
#pragma unroll
            for (int mi = 0; mi < 4; ++mi)
#pragma unroll
                for (int ni = 0; ni < 4; ++ni)
                    acc[mi][ni] = __builtin_amdgcn_wmma_f32_16x16x32_bf16(
                        false, a[mi].v, false, b[ni].v, (short)0, acc[mi][ni], false, false);
        }
    }

#pragma unroll
    for (int mi = 0; mi < 4; ++mi) {
#pragma unroll
        for (int ni = 0; ni < 4; ++ni) {
            const int col  = blockN + waveN * 64 + ni * 16 + lx;
            const int row0 = blockM + waveM * 64 + mi * 16 + half * 8;
#pragma unroll
            for (int r = 0; r < 8; ++r) {
                size_t idx = (size_t)(row0 + r) * Nout + col;
                float v = acc[mi][ni][r];
                if (resid) v += resid[idx];
                C[idx] = v;
            }
        }
    }
}

// ---------------------------------------------------------------------------
// Per-head q/k RMSNorm + RoPE; reshuffle into [n,h,T,64] (Q,K) and V^T
// [n,h,64,T] bf16. One thread per (n,t,h).
// ---------------------------------------------------------------------------
__global__ __launch_bounds__(256) void qkrope_kernel(
    const float* __restrict__ qkv, bf16* __restrict__ Q,
    bf16* __restrict__ Kb, bf16* __restrict__ Vt)
{
    int idx = blockIdx.x * 256 + threadIdx.x;
    if (idx >= NB * TT * NHEAD) return;
    const int h = idx % NHEAD;
    const int t = (idx / NHEAD) % TT;
    const int n = idx / (NHEAD * TT);

    const float* rowp = qkv + (size_t)(n * TT + t) * (3 * DDIM);
    const float* qp = rowp + h * HDIM;
    const float* kp = rowp + (NHEAD + h) * HDIM;
    const float* vp = rowp + (2 * NHEAD + h) * HDIM;

    float qv[HDIM], kv[HDIM];
    float qss = 0.f, kss = 0.f;
#pragma unroll
    for (int i = 0; i < HDIM; ++i) {
        qv[i] = qp[i]; qss += qv[i] * qv[i];
        kv[i] = kp[i]; kss += kv[i] * kv[i];
    }
    const float qsc = rsqrtf(qss / (float)HDIM + EPSF);
    const float ksc = rsqrtf(kss / (float)HDIM + EPSF);

    bf16* Qh = Q  + ((size_t)(n * NHEAD + h) * TT + t) * HDIM;
    bf16* Kh = Kb + ((size_t)(n * NHEAD + h) * TT + t) * HDIM;
#pragma unroll
    for (int i = 0; i < HDIM / 2; ++i) {
        // freqs = (1/BASE)^linspace(0,1,16), zero-padded to 32
        float freq = (i < HDIM / 4) ? __powf(10000.0f, -(float)i / 15.0f) : 0.0f;
        float th = (float)t * freq;
        float cs = __cosf(th), sn = __sinf(th);
        float q1 = qv[i] * qsc, q2 = qv[i + HDIM / 2] * qsc;
        float k1 = kv[i] * ksc, k2 = kv[i + HDIM / 2] * ksc;
        Qh[i]            = to_bf16(q1 * cs + q2 * sn);
        Qh[i + HDIM / 2] = to_bf16(-q1 * sn + q2 * cs);
        Kh[i]            = to_bf16(k1 * cs + k2 * sn);
        Kh[i + HDIM / 2] = to_bf16(-k1 * sn + k2 * cs);
    }
    bf16* Vh = Vt + (size_t)(n * NHEAD + h) * HDIM * TT;
#pragma unroll
    for (int d = 0; d < HDIM; ++d)
        Vh[(size_t)d * TT + t] = to_bf16(vp[d]);
}

// ---------------------------------------------------------------------------
// Sliding-window attention. Block = 2 waves = 32 queries of one (n,h,chunk).
// Wave computes 16x256 masked scores (WMMA), softmax via LDS, then 16x64
// output (WMMA over bf16 probs), stored as bf16 rows [n*T+t, h*64+d].
// ---------------------------------------------------------------------------
__global__ __launch_bounds__(64) void attn_kernel(
    const bf16* __restrict__ Q, const bf16* __restrict__ Kb,
    const bf16* __restrict__ Vt, bf16* __restrict__ O)
{
    __shared__ float sS[2][16][2 * WW];    // f32 scores -> unnormalized probs
    __shared__ bf16  sP[2][16][2 * WW];    // normalized probs (A-operand)
    __shared__ float sRed[2][2][16];

    const int lane = threadIdx.x & 31;
    const int w    = threadIdx.x >> 5;
    const int half = lane >> 4;
    const int lx   = lane & 15;

    const int strip = blockIdx.x;          // 0..3 -> 32 queries each
    const int c     = blockIdx.y;          // chunk
    const int nh    = blockIdx.z;          // n*NH + h
    const int n = nh / NHEAD, h = nh % NHEAD;

    const bf16* Qh = Q  + (size_t)nh * TT * HDIM;
    const bf16* Kh = Kb + (size_t)nh * TT * HDIM;
    const bf16* Vh = Vt + (size_t)nh * HDIM * TT;

    const int qloc0 = strip * 32 + w * 16;   // query index within chunk
    const int qtok0 = c * WW + qloc0;
    const int ktok0 = c * WW - WW;           // window start token

    // Q fragments for K-dim 0..31 and 32..63
    FragB aq0, aq1;
    {
        const bf16* qr = Qh + (size_t)(qtok0 + lx) * HDIM + half * 8;
        aq0.q[0] = *(const uint4*)(qr);
        aq0.q[1] = *(const uint4*)(qr + 16);
        aq1.q[0] = *(const uint4*)(qr + 32);
        aq1.q[1] = *(const uint4*)(qr + 48);
    }

    // ---- scores: 16 key tiles ----
    for (int j = 0; j < 16; ++j) {
        int kt = ktok0 + j * 16 + lx;
        if (kt < 0) kt = 0;                  // masked anyway (c==0, m<W)
        const bf16* kr = Kh + (size_t)kt * HDIM + half * 8;
        FragB b0, b1;
        b0.q[0] = *(const uint4*)(kr);
        b0.q[1] = *(const uint4*)(kr + 16);
        b1.q[0] = *(const uint4*)(kr + 32);
        b1.q[1] = *(const uint4*)(kr + 48);
        v8f cc = {};
        cc = __builtin_amdgcn_wmma_f32_16x16x32_bf16(
            false, aq0.v, false, b0.v, (short)0, cc, false, false);
        cc = __builtin_amdgcn_wmma_f32_16x16x32_bf16(
            false, aq1.v, false, b1.v, (short)0, cc, false, false);
        const int m = j * 16 + lx;
#pragma unroll
        for (int r = 0; r < 8; ++r) {
            const int qrow = half * 8 + r;
            const int jloc = qloc0 + qrow;
            const bool ok = (m > jloc) && (m <= WW + jloc) && ((c > 0) || (m >= WW));
            sS[w][qrow][m] = ok ? cc[r] * 0.125f : -3.0e38f;
        }
    }
    __syncthreads();

    // ---- softmax: 2 lanes per row (lane lx handles segment 'half') ----
    {
        const int row = lx, seg = half;
        float mx = -3.0e38f;
        for (int m = seg * WW; m < seg * WW + WW; ++m)
            mx = fmaxf(mx, sS[w][row][m]);
        sRed[w][seg][row] = mx;
        __syncthreads();
        mx = fmaxf(sRed[w][0][row], sRed[w][1][row]);
        __syncthreads();
        float sum = 0.f;
        for (int m = seg * WW; m < seg * WW + WW; ++m) {
            float p = __expf(sS[w][row][m] - mx);
            sS[w][row][m] = p;
            sum += p;
        }
        sRed[w][seg][row] = sum;
        __syncthreads();
        const float rs = 1.0f / (sRed[w][0][row] + sRed[w][1][row]);
        for (int m = seg * WW; m < seg * WW + WW; ++m)
            sP[w][row][m] = to_bf16(sS[w][row][m] * rs);
    }
    __syncthreads();

    // ---- out = P @ V : 4 d-tiles, K=256 in 8 WMMA steps ----
    for (int t = 0; t < 4; ++t) {
        v8f acc = {};
        for (int s8 = 0; s8 < 8; ++s8) {
            const int base = s8 * 32 + half * 8;
            FragB a, b;
            a.q[0] = *(const uint4*)(&sP[w][lx][base]);
            a.q[1] = *(const uint4*)(&sP[w][lx][base + 16]);
            int t0 = ktok0 + base;
            int t1 = t0 + 16;
            if (t0 < 0) t0 = 0;              // probs are 0 there
            if (t1 < 0) t1 = 0;
            const bf16* vr = Vh + (size_t)(t * 16 + lx) * TT;
            b.q[0] = *(const uint4*)(vr + t0);
            b.q[1] = *(const uint4*)(vr + t1);
            acc = __builtin_amdgcn_wmma_f32_16x16x32_bf16(
                false, a.v, false, b.v, (short)0, acc, false, false);
        }
        const int dcol = h * HDIM + t * 16 + lx;
#pragma unroll
        for (int r = 0; r < 8; ++r) {
            const int tok = qtok0 + half * 8 + r;
            O[(size_t)(n * TT + tok) * DDIM + dcol] = to_bf16(acc[r]);
        }
    }
}

// ---------------------------------------------------------------------------
extern "C" void kernel_launch(void* const* d_in, const int* in_sizes, int n_in,
                              void* d_out, int out_size, void* d_ws, size_t ws_size,
                              hipStream_t stream)
{
    const float* input = (const float*)d_in[0];   // [N,T,D]
    const float* qkvw  = (const float*)d_in[1];   // [3,D,D]
    const float* ow    = (const float*)d_in[2];   // [D,D]
    const float* osc   = (const float*)d_in[3];   // [D]
    float* out = (float*)d_out;                   // [N,T,D] f32

    char* p = (char*)d_ws;
    auto carve = [&](size_t bytes) {
        char* r = p;
        p += (bytes + 255) & ~(size_t)255;
        return (void*)r;
    };
    bf16*  xn   = (bf16*) carve((size_t)MROWS * DDIM * 2);
    bf16*  Wq   = (bf16*) carve((size_t)3 * DDIM * DDIM * 2);
    bf16*  W2   = (bf16*) carve((size_t)DDIM * DDIM * 2);
    float* qkv  = (float*)carve((size_t)MROWS * 3 * DDIM * 4);
    bf16*  Qb   = (bf16*) carve((size_t)NB * NHEAD * TT * HDIM * 2);
    bf16*  Kbuf = (bf16*) carve((size_t)NB * NHEAD * TT * HDIM * 2);
    bf16*  Vt   = (bf16*) carve((size_t)NB * NHEAD * TT * HDIM * 2);
    bf16*  attn = (bf16*) carve((size_t)MROWS * DDIM * 2);

    // 1. weights -> bf16 (o_scale folded)
    {
        int total = 3 * DDIM * DDIM + DDIM * DDIM;
        convw_kernel<<<(total + 255) / 256, 256, 0, stream>>>(qkvw, ow, osc, Wq, W2);
    }
    // 2. input RMSNorm -> bf16
    rmsnorm_kernel<<<MROWS, 256, 0, stream>>>(input, xn);
    // 3. QKV projection: [8192,1024] x [3072,1024]^T, 128x256 block tiles
    gemm_bf16_kernel<<<dim3(3 * DDIM / 256, MROWS / 128), 256, 0, stream>>>(
        xn, Wq, qkv, nullptr, MROWS, 3 * DDIM, DDIM);
    // 4. head RMS + RoPE + layout shuffle
    qkrope_kernel<<<(NB * TT * NHEAD + 255) / 256, 256, 0, stream>>>(qkv, Qb, Kbuf, Vt);
    // 5. sliding-window attention
    attn_kernel<<<dim3(WW / 32, CCHUNK, NB * NHEAD), 64, 0, stream>>>(Qb, Kbuf, Vt, attn);
    // 6. O-proj + residual -> f32 output
    gemm_bf16_kernel<<<dim3(DDIM / 256, MROWS / 128), 256, 0, stream>>>(
        attn, W2, out, input, MROWS, DDIM, DDIM);
}
